// CrystalConvLayer_67190468378979
// MI455X (gfx1250) — compile-verified, matrix operand order checked by reading, not run on hardware
//
#include <hip/hip_runtime.h>
#include <hip/hip_bf16.h>

typedef __attribute__((ext_vector_type(16))) _Float16 v16h;
typedef __attribute__((ext_vector_type(8)))  float    v8f;
typedef __attribute__((ext_vector_type(4)))  unsigned int u32x4;
typedef __attribute__((ext_vector_type(8)))  int      i32x8;
typedef __attribute__((ext_vector_type(4)))  int      i32x4;

#define NN      100000
#define EE      1000000
#define HH      128
#define EAF     48
#define SHF     4
#define TRUNCF  50
#define NT8     8          // 128 cols / 16

// k position inside a 32-wide k-tile for half element j (0..15) of a lane,
// matching the CDNA5 16-bit A-matrix 16x32 layout (and mirrored for B).
__device__ __forceinline__ int kmap(int j, int half) {
    return (j & 7) + ((j & 8) << 1) + (half ? 8 : 0);
}

__device__ __forceinline__ v8f wmma_f16(v16h a, v16h b, v8f c) {
    return __builtin_amdgcn_wmma_f32_16x16x32_f16(false, a, false, b, (short)0, c, false, false);
}

__device__ __forceinline__ float silu(float x) {
    return x / (1.0f + __expf(-x));
}

// ---------------------------------------------------------------------------
// Stage a contiguous weight-fragment blob (global -> LDS), shared by all 8
// waves of the block.  Primary path: Tensor Data Mover, one 1-D descriptor
// (n_dwords <= 65535 elements of 4B), issued by wave 0, completion via
// TENSORcnt.  Fallback: cooperative b128 copies.
// ---------------------------------------------------------------------------
__device__ __forceinline__ void stage_weights(_Float16* lds_dst,
                                              const _Float16* gsrc,
                                              int halves) {
#if __has_builtin(__builtin_amdgcn_tensor_load_to_lds)
    if ((threadIdx.x >> 5) == 0) {
        const unsigned e  = (unsigned)(halves >> 1);          // dwords
        const unsigned la = (unsigned)(uintptr_t)lds_dst;     // LDS byte offset
        const unsigned long long ga = (unsigned long long)(uintptr_t)gsrc;

        u32x4 g0;
        g0[0] = 1u;                                           // count=1, user D#
        g0[1] = la;                                           // lds_addr
        g0[2] = (unsigned)ga;                                 // global_addr[31:0]
        g0[3] = ((unsigned)(ga >> 32) & 0x01FFFFFFu)          // global_addr[56:32]
                | (2u << 30);                                 // type=2 (image)
        i32x8 g1;
        g1[0] = (int)(2u << 16);                              // data_size = 4B
        g1[1] = (int)((e & 0xFFFFu) << 16);                   // tensor_dim0[15:0]
        g1[2] = (int)((e >> 16) | (1u << 16));                // tensor_dim0[31:16], tensor_dim1=1
        g1[3] = (int)((e & 0xFFFFu) << 16);                   // tile_dim0 = e
        g1[4] = 1;                                            // tile_dim1 = 1
        g1[5] = (int)e;                                       // tensor_dim0_stride
        g1[6] = 0;
        g1[7] = 0;
        i32x4 g2 = {0, 0, 0, 0};
        i32x4 g3 = {0, 0, 0, 0};
#if __clang_major__ >= 23
        i32x8 g4 = {0, 0, 0, 0, 0, 0, 0, 0};
        __builtin_amdgcn_tensor_load_to_lds(g0, g1, g2, g3, g4, 0);
#else
        __builtin_amdgcn_tensor_load_to_lds(g0, g1, g2, g3, 0);
#endif
        __builtin_amdgcn_s_wait_tensorcnt(0);
    }
#else
    {
        const uint4* s = (const uint4*)gsrc;
        uint4*       d = (uint4*)lds_dst;
        const int n16 = halves >> 3;                          // 16B chunks
        for (int i = threadIdx.x; i < n16; i += blockDim.x)
            d[i] = s[i];
    }
#endif
    __syncthreads();
}

// ---------------------------------------------------------------------------
// Repack the four weight matrices (row-major [K][128] f32) into f16 B-fragment
// layout: frag[(kt*8+nt)*32 + lane][j], j=0..15, k = kt*32 + kmap(j, lane>>4),
// n = nt*16 + (lane&15).  K padded with zeros (50 -> 64).
// ---------------------------------------------------------------------------
__global__ void prep_weights_kernel(const float* __restrict__ W1e,
                                    const float* __restrict__ W2e,
                                    const float* __restrict__ W1n,
                                    const float* __restrict__ W2n,
                                    _Float16* __restrict__ f1e,   //  8192 halves (KT=2)
                                    _Float16* __restrict__ f2e,   // 16384 halves (KT=4)
                                    _Float16* __restrict__ f1n,   // 32768 halves (KT=8)
                                    _Float16* __restrict__ f2n) { // 16384 halves (KT=4)
    const int total = 8192 + 16384 + 32768 + 16384;
    for (int t = blockIdx.x * blockDim.x + threadIdx.x; t < total;
         t += gridDim.x * blockDim.x) {
        const float* W; _Float16* F; int Kact, rel;
        if (t < 8192)        { W = W1e; F = f1e; Kact = TRUNCF; rel = t; }
        else if (t < 24576)  { W = W2e; F = f2e; Kact = 128;    rel = t - 8192; }
        else if (t < 57344)  { W = W1n; F = f1n; Kact = 256;    rel = t - 24576; }
        else                 { W = W2n; F = f2n; Kact = 128;    rel = t - 57344; }
        int j    = rel & 15;
        int lane = (rel >> 4) & 31;
        int ktnt = rel >> 9;
        int nt   = ktnt & 7;
        int kt   = ktnt >> 3;
        int n    = nt * 16 + (lane & 15);
        int k    = kt * 32 + kmap(j, lane >> 4);
        float v  = (k < Kact) ? W[k * HH + n] : 0.0f;
        F[rel] = (_Float16)v;
    }
}

// ---------------------------------------------------------------------------
// Edge phase: per wave, 16 edges.
//   t1     = silu(edge_input[16,50] @ W1e + b1e)     (2 k-tiles,  8 n-tiles)
//   w_edge = t1 @ W2e + b2e                          (4 k-tiles,  8 n-tiles)
//   agg[dst] += h[src] * w_edge   (f32 global atomics, L2-resident)
// Weights staged in LDS by TDM (48 KB), shared by all 8 waves.
// ---------------------------------------------------------------------------
__global__ void edge_kernel(const float* __restrict__ h,
                            const int*   __restrict__ ei,
                            const float* __restrict__ ea,
                            const float* __restrict__ esh,
                            const float* __restrict__ b1e,
                            const float* __restrict__ b2e,
                            const _Float16* __restrict__ fedge,  // f1e|f2e contiguous
                            float* __restrict__ agg) {
    __shared__ _Float16 wfrag[8192 + 16384];          // 48 KB weight fragments
    __shared__ _Float16 t1s[8][16 * HH];              // 32 KB activation tiles

    stage_weights(wfrag, fedge, 8192 + 16384);

    const int wave = threadIdx.x >> 5;
    const int lane = threadIdx.x & 31;
    const int half = lane >> 4;
    const int mrow = lane & 15;
    const int e0   = (blockIdx.x * 8 + wave) * 16;
    const int e_m  = e0 + mrow;                       // A-matrix row -> edge
    const bool mvalid = (e_m < EE);

    // --- A fragments for GEMM1 (K = 50 padded to 64) -----------------------
    v16h a[2];
    #pragma unroll
    for (int kt = 0; kt < 2; ++kt) {
        #pragma unroll
        for (int j = 0; j < 16; ++j) {
            int k = kt * 32 + kmap(j, half);
            float v = 0.0f;
            if (mvalid && k < TRUNCF)
                v = (k < EAF) ? ea[e_m * EAF + k] : esh[e_m * SHF + (k - EAF)];
            a[kt][j] = (_Float16)v;
        }
    }

    // --- GEMM1 + bias + silu, write t1 to LDS ------------------------------
    const v16h* fb1 = (const v16h*)wfrag;
    #pragma unroll
    for (int nt = 0; nt < NT8; ++nt) {
        v8f c = {};
        #pragma unroll
        for (int kt = 0; kt < 2; ++kt)
            c = wmma_f16(a[kt], fb1[(kt * NT8 + nt) * 32 + lane], c);
        float bb = b1e[nt * 16 + mrow];               // bias depends on column n
        #pragma unroll
        for (int r = 0; r < 8; ++r) {
            float x = c[r] + bb;
            // C element (m = r + 8*half, n = nt*16 + mrow)
            t1s[wave][(r + 8 * half) * HH + nt * 16 + mrow] = (_Float16)silu(x);
        }
    }
    __syncthreads();

    // --- A fragments for GEMM2 (K = 128) -----------------------------------
    v16h a2[4];
    #pragma unroll
    for (int kt = 0; kt < 4; ++kt)
        #pragma unroll
        for (int j = 0; j < 16; ++j)
            a2[kt][j] = t1s[wave][mrow * HH + kt * 32 + kmap(j, half)];

    // --- src/dst for the 8 rows this lane owns in the C layout -------------
    int se[8], de[8];
    #pragma unroll
    for (int r = 0; r < 8; ++r) {
        int e = e0 + r + 8 * half;
        bool v = (e < EE);
        se[r] = v ? ei[e] : 0;
        de[r] = v ? ei[EE + e] : 0;
    }

    // --- GEMM2 + bias, gather-multiply-scatter ------------------------------
    const v16h* fb2 = (const v16h*)(wfrag + 8192);
    #pragma unroll
    for (int nt = 0; nt < NT8; ++nt) {
        v8f c = {};
        #pragma unroll
        for (int kt = 0; kt < 4; ++kt)
            c = wmma_f16(a2[kt], fb2[(kt * NT8 + nt) * 32 + lane], c);
        float bb = b2e[nt * 16 + mrow];
        int col = nt * 16 + mrow;
        #pragma unroll
        for (int r = 0; r < 8; ++r) {
            int e = e0 + r + 8 * half;
            if (e < EE) {
                float w   = c[r] + bb;
                float msg = h[se[r] * HH + col] * w;
                atomicAdd(&agg[de[r] * HH + col], msg);
            }
        }
    }
}

// ---------------------------------------------------------------------------
// Node phase: per wave, 16 nodes.
//   t1    = silu([h | agg][16,256] @ W1n + b1n)      (8 k-tiles)
//   h_upd = t1 @ W2n + b2n                           (4 k-tiles)
//   out   = LayerNorm(h + h_upd) * g + b             (shfl row reduction)
// Weights staged in LDS by TDM (96 KB), shared by all 8 waves.
// ---------------------------------------------------------------------------
__global__ void node_kernel(const float* __restrict__ h,
                            const float* __restrict__ agg,
                            const float* __restrict__ b1n,
                            const float* __restrict__ b2n,
                            const _Float16* __restrict__ fnode, // f1n|f2n contiguous
                            const float* __restrict__ ln_g,
                            const float* __restrict__ ln_b,
                            float* __restrict__ out) {
    __shared__ _Float16 wfrag[32768 + 16384];         // 96 KB weight fragments
    __shared__ _Float16 t1s[8][16 * HH];              // 32 KB activation tiles

    stage_weights(wfrag, fnode, 32768 + 16384);

    const int wave = threadIdx.x >> 5;
    const int lane = threadIdx.x & 31;
    const int half = lane >> 4;
    const int mrow = lane & 15;
    const int n0   = (blockIdx.x * 8 + wave) * 16;
    const int nd_m = n0 + mrow;
    const bool mvalid = (nd_m < NN);

    // --- A fragments for GEMM1 over [h | agg], K = 256 ----------------------
    v16h a[8];
    #pragma unroll
    for (int kt = 0; kt < 8; ++kt) {
        #pragma unroll
        for (int j = 0; j < 16; ++j) {
            int k = kt * 32 + kmap(j, half);
            float v = 0.0f;
            if (mvalid)
                v = (k < HH) ? h[nd_m * HH + k] : agg[nd_m * HH + (k - HH)];
            a[kt][j] = (_Float16)v;
        }
    }

    const v16h* fb1 = (const v16h*)wfrag;
    #pragma unroll
    for (int nt = 0; nt < NT8; ++nt) {
        v8f c = {};
        #pragma unroll
        for (int kt = 0; kt < 8; ++kt)
            c = wmma_f16(a[kt], fb1[(kt * NT8 + nt) * 32 + lane], c);
        float bb = b1n[nt * 16 + mrow];
        #pragma unroll
        for (int r = 0; r < 8; ++r)
            t1s[wave][(r + 8 * half) * HH + nt * 16 + mrow] = (_Float16)silu(c[r] + bb);
    }
    __syncthreads();

    v16h a2[4];
    #pragma unroll
    for (int kt = 0; kt < 4; ++kt)
        #pragma unroll
        for (int j = 0; j < 16; ++j)
            a2[kt][j] = t1s[wave][mrow * HH + kt * 32 + kmap(j, half)];

    // --- GEMM2 + bias + residual, kept in registers -------------------------
    const v16h* fb2 = (const v16h*)(wfrag + 32768);
    v8f xv[NT8];
    #pragma unroll
    for (int nt = 0; nt < NT8; ++nt) {
        v8f c = {};
        #pragma unroll
        for (int kt = 0; kt < 4; ++kt)
            c = wmma_f16(a2[kt], fb2[(kt * NT8 + nt) * 32 + lane], c);
        float bb = b2n[nt * 16 + mrow];
        int col = nt * 16 + mrow;
        #pragma unroll
        for (int r = 0; r < 8; ++r) {
            int node = n0 + r + 8 * half;
            float hres = (node < NN) ? h[node * HH + col] : 0.0f;
            c[r] = hres + c[r] + bb;
        }
        xv[nt] = c;
    }

    // --- LayerNorm: row m = r + 8*half lives in one 16-lane half-wave -------
    float mean_r[8], inv_r[8];
    #pragma unroll
    for (int r = 0; r < 8; ++r) {
        float s1 = 0.0f, s2 = 0.0f;
        #pragma unroll
        for (int nt = 0; nt < NT8; ++nt) { float v = xv[nt][r]; s1 += v; s2 += v * v; }
        #pragma unroll
        for (int m = 1; m < 16; m <<= 1) {
            s1 += __shfl_xor(s1, m, 16);
            s2 += __shfl_xor(s2, m, 16);
        }
        float mu  = s1 * (1.0f / HH);
        float var = s2 * (1.0f / HH) - mu * mu;
        mean_r[r] = mu;
        inv_r[r]  = rsqrtf(var + 1e-5f);
    }

    #pragma unroll
    for (int nt = 0; nt < NT8; ++nt) {
        int col = nt * 16 + mrow;
        float g = ln_g[col], b = ln_b[col];
        #pragma unroll
        for (int r = 0; r < 8; ++r) {
            int node = n0 + r + 8 * half;
            if (node < NN)
                out[node * HH + col] = (xv[nt][r] - mean_r[r]) * inv_r[r] * g + b;
        }
    }
}

// ---------------------------------------------------------------------------
extern "C" void kernel_launch(void* const* d_in, const int* in_sizes, int n_in,
                              void* d_out, int out_size, void* d_ws, size_t ws_size,
                              hipStream_t stream) {
    const float* h    = (const float*)d_in[0];
    const int*   ei   = (const int*)  d_in[1];
    const float* ea   = (const float*)d_in[2];
    const float* esh  = (const float*)d_in[3];
    const float* W1e  = (const float*)d_in[4];
    const float* b1e  = (const float*)d_in[5];
    const float* W2e  = (const float*)d_in[6];
    const float* b2e  = (const float*)d_in[7];
    const float* W1n  = (const float*)d_in[8];
    const float* b1n  = (const float*)d_in[9];
    const float* W2n  = (const float*)d_in[10];
    const float* b2n  = (const float*)d_in[11];
    const float* ln_g = (const float*)d_in[12];
    const float* ln_b = (const float*)d_in[13];
    float* out = (float*)d_out;

    // workspace layout (fragment blobs kept contiguous per phase for TDM)
    char* ws = (char*)d_ws;
    float*    agg = (float*)ws;                          // N*H*4 = 51,200,000 B
    size_t off = (size_t)NN * HH * sizeof(float);
    _Float16* f1e = (_Float16*)(ws + off); off +=  8192 * sizeof(_Float16);
    _Float16* f2e = (_Float16*)(ws + off); off += 16384 * sizeof(_Float16);
    _Float16* f1n = (_Float16*)(ws + off); off += 32768 * sizeof(_Float16);
    _Float16* f2n = (_Float16*)(ws + off);

    hipMemsetAsync(agg, 0, (size_t)NN * HH * sizeof(float), stream);

    prep_weights_kernel<<<288, 256, 0, stream>>>(W1e, W2e, W1n, W2n,
                                                 f1e, f2e, f1n, f2n);

    int edge_blocks = (EE + 127) / 128;                  // 8 waves * 16 edges
    edge_kernel<<<edge_blocks, 256, 0, stream>>>(h, ei, ea, esh, b1e, b2e,
                                                 f1e, agg);

    int node_blocks = (NN + 127) / 128;
    node_kernel<<<node_blocks, 256, 0, stream>>>(h, agg, b1n, b2n,
                                                 f1n, ln_g, ln_b, out);
}
